// TextGenerationModel_84954453115647
// MI455X (gfx1250) — matrix operand order repaired and checked by reading, not compiled, not run on hardware
//
#include <hip/hip_runtime.h>
#include <math.h>

// ---------------------------------------------------------------------------
// Types for CDNA5 WMMA (wave32, v_wmma_f32_16x16x32_bf16)
// ---------------------------------------------------------------------------
typedef __bf16 bf16;
typedef __attribute__((ext_vector_type(16))) __bf16 v16bf;
typedef __attribute__((ext_vector_type(8)))  __bf16 bf16x8;
typedef __attribute__((ext_vector_type(8)))  float  v8f;

#define DEV __device__ __forceinline__

// Problem constants (from reference): B=128, S=1024, V=128, H=256
static constexpr int    kB = 128;
static constexpr int    kS = 1024;
static constexpr int    kV = 128;
static constexpr int    kH = 256;
static constexpr int    kG = 4 * kH;          // 1024 gates
static constexpr long   kM = (long)kB * kS;   // 131072 rows

// ---------------------------------------------------------------------------
// Fragment loaders per CDNA5 ISA VGPR layouts (CDNA5 ISA §7.12.2)
// A (16x32 bf16, MxK): lane L -> row = L&15 ; element i -> k = 16*(i/8) + 8*hi + (i%8)
// B (32x16 bf16, KxN): lane L -> col = L&15 ; element i -> k = 16*(i/8) + 8*hi + (i%8) (same as A)
// C/D (16x16 f32):     reg r  -> row = r + 8*hi ; col = L&15
// ---------------------------------------------------------------------------
DEV v16bf load_a_frag_bf16(const bf16* __restrict__ p, int hi) {
  bf16x8 lo = *(const bf16x8*)(p + hi * 8);        // k = 8*hi .. 8*hi+7
  bf16x8 hh = *(const bf16x8*)(p + 16 + hi * 8);   // k = 16+8*hi .. 16+8*hi+7
  return __builtin_shufflevector(lo, hh, 0, 1, 2, 3, 4, 5, 6, 7,
                                 8, 9, 10, 11, 12, 13, 14, 15);
}

DEV v16bf load_a_frag_f32(const float* __restrict__ p, int hi) {
  const float4* q0 = (const float4*)(p + hi * 8);
  const float4* q1 = (const float4*)(p + 16 + hi * 8);
  float4 f0 = q0[0], f1 = q0[1], f2 = q1[0], f3 = q1[1];
  float t[16] = {f0.x, f0.y, f0.z, f0.w, f1.x, f1.y, f1.z, f1.w,
                 f2.x, f2.y, f2.z, f2.w, f3.x, f3.y, f3.z, f3.w};
  v16bf a;
#pragma unroll
  for (int i = 0; i < 16; ++i) a[i] = (bf16)t[i];
  return a;
}

DEV v16bf load_a_frag(const bf16* p, int hi) { return load_a_frag_bf16(p, hi); }
DEV v16bf load_a_frag(const float* p, int hi) { return load_a_frag_f32(p, hi); }

DEV v16bf load_b_frag(const bf16* __restrict__ rowp, int hi) {
  // k = 8*hi .. 8*hi+7, then 16+8*hi .. 16+8*hi+7 (matches A fragment)
  bf16x8 lo = *(const bf16x8*)(rowp + hi * 8);
  bf16x8 hh = *(const bf16x8*)(rowp + 16 + hi * 8);
  return __builtin_shufflevector(lo, hh, 0, 1, 2, 3, 4, 5, 6, 7,
                                 8, 9, 10, 11, 12, 13, 14, 15);
}

DEV float fast_sigmoid(float x) { return 1.0f / (1.0f + __expf(-x)); }
DEV float fast_tanh(float x) {
  float e = __expf(-2.0f * fabsf(x));
  float r = (1.0f - e) / (1.0f + e);
  return copysignf(r, x);
}

// ---------------------------------------------------------------------------
// GEMM: C[M,N] = A[M,K] * W[N,K]^T + bias[N]    (A fp32 or bf16, C fp32 or bf16)
// Block = 256 threads = 8 waves arranged 4(M) x 2(N); block tile 64M x 128N.
// Each wave: 16M x 64N via 4 accumulators; K stepped by 32.
// grid = (N/128, M/64)
// ---------------------------------------------------------------------------
template <typename AT, typename OT>
__global__ __launch_bounds__(256) void lstm_gemm_wmma(
    const AT* __restrict__ A, const bf16* __restrict__ W,
    const float* __restrict__ bias, OT* __restrict__ C,
    int N, int K) {
  const int lane = threadIdx.x & 31;
  const int wave = threadIdx.x >> 5;
  const int wm = wave >> 1;          // 0..3  (M)
  const int wn = wave & 1;           // 0..1  (N)
  const int hi = (lane >> 4) & 1;
  const int l15 = lane & 15;

  const long arow = (long)blockIdx.y * 64 + wm * 16 + l15;
  const int  n0   = blockIdx.x * 128 + wn * 64;   // 4 n-tiles: n0 + 16q

  v8f acc[4] = {};
  const AT* aptr = A + arow * (long)K;

  for (int k0 = 0; k0 < K; k0 += 32) {
    v16bf af = load_a_frag(aptr + k0, hi);
    if (k0 + 32 < K)  // hint next W rows into cache (global_prefetch_b8)
      __builtin_prefetch((const void*)(W + (long)(n0 + l15) * K + k0 + 32), 0, 0);
#pragma unroll
    for (int q = 0; q < 4; ++q) {
      const bf16* bp = W + (long)(n0 + q * 16 + l15) * K + k0;
      v16bf bf = load_b_frag(bp, hi);
      acc[q] = __builtin_amdgcn_wmma_f32_16x16x32_bf16(
          false, af, false, bf, (short)0, acc[q], false, false);
    }
  }

  const long crow = (long)blockIdx.y * 64 + wm * 16 + hi * 8;
#pragma unroll
  for (int q = 0; q < 4; ++q) {
    const int n = n0 + q * 16 + l15;
    const float bv = bias[n];
#pragma unroll
    for (int r = 0; r < 8; ++r) {
      float v = acc[q][r] + bv;
      C[(crow + r) * (long)N + n] = (OT)v;
    }
  }
}

// ---------------------------------------------------------------------------
// Sequential LSTM scan for one layer.
// grid = B/16 blocks; block = 512 threads = 16 waves.
// Each block owns 16 batch rows; h (16x256 bf16) lives in LDS, c in registers.
// Per step: gates[16,1024] = h @ Whh^T (+xg), WMMA over K=256, then pointwise.
// Wave w covers gate columns j = 16w + (lane&15); its 4 accumulators (q=0..3)
// are exactly the i,f,g,o gates for column j (n = j + 256q).
//
// * W_hh fragments are re-read from global (L2-resident, 512 KB) every step;
//   the opaque integer offset blocks LICM without destroying the pointer's
//   address space (keeps global_load_b128, not flat), so no spills.
// * The kt loop is manually double-buffered: loads for kt+1 issue before the
//   4 WMMAs of kt, so waits never fully drain LOADcnt.
// * xg(t) is staged into LDS with GLOBAL_LOAD_ASYNC_TO_LDS_B128 (ASYNCcnt),
//   issued at step start and drained just before the mid-step barrier.
// ---------------------------------------------------------------------------
__global__ __launch_bounds__(512) void lstm_scan_wmma(
    const bf16* __restrict__ xg,    // [B, S, 4H] bf16 (bias already folded in)
    const bf16* __restrict__ Whh,   // [4H, H] bf16
    bf16* __restrict__ hout,        // [B, S, H] bf16
    int S) {
  constexpr int H = kH, G = kG;
  constexpr int HP = H + 8;                 // +16B pad  -> conflict-free ds_load
  constexpr int GP = G + 16;                // +32B pad  -> conflict-free gate reads
  __shared__ bf16 hbuf[16][HP];
  __shared__ bf16 xgbuf[16][GP];

  const int tid  = threadIdx.x;
  const int lane = tid & 31;
  const int wave = tid >> 5;                // 0..15
  const int hi   = (lane >> 4) & 1;
  const int l15  = lane & 15;
  const int b0   = blockIdx.x * 16;
  const int j    = wave * 16 + l15;         // h / gate column 0..255

  // xg staging: wave w copies batch-row w of the [16 x 1024] bf16 step tile;
  // each lane moves 64 contiguous bytes via 4 async b128 transfers.
  const int xrow = wave;
  const unsigned voff = (unsigned)(lane * 64);              // global byte offset
  const unsigned ldsa =
      (unsigned)(size_t)(&xgbuf[xrow][lane * 32]);          // LDS addr = low 32b

  float c[8];
#pragma unroll
  for (int r = 0; r < 8; ++r) c[r] = 0.0f;

  for (int idx = tid; idx < 16 * HP; idx += 512)
    (&hbuf[0][0])[idx] = (bf16)0.0f;
  __syncthreads();

  long wofs = 0;  // opaque 0: defeats LICM of W fragment loads
  for (int t = 0; t < S; ++t) {
    asm volatile("" : "+s"(wofs));
    const bf16* Wt = Whh + wofs;

    // Kick off async global->LDS staging of xg(t); overlaps with the GEMM.
    {
      const bf16* srow = xg + ((long)(b0 + xrow) * S + t) * G;
      asm volatile(
          "global_load_async_to_lds_b128 %0, %1, %2 offset:0\n\t"
          "global_load_async_to_lds_b128 %0, %1, %2 offset:16\n\t"
          "global_load_async_to_lds_b128 %0, %1, %2 offset:32\n\t"
          "global_load_async_to_lds_b128 %0, %1, %2 offset:48"
          :: "v"(ldsa), "v"(voff), "s"(srow)
          : "memory");
    }

    // ---- gates = h(t-1) @ Whh^T, double-buffered over kt -------------------
    v8f acc[4] = {};
    const bf16* wrow = Wt + (long)j * H;    // row j; +256*H per gate block q
    v16bf af = load_a_frag_bf16(&hbuf[l15][0], hi);
    v16bf bq0 = load_b_frag(wrow + 0L * 256 * H, hi);
    v16bf bq1 = load_b_frag(wrow + 1L * 256 * H, hi);
    v16bf bq2 = load_b_frag(wrow + 2L * 256 * H, hi);
    v16bf bq3 = load_b_frag(wrow + 3L * 256 * H, hi);
#pragma unroll
    for (int kt = 0; kt < 8; ++kt) {
      v16bf afn = af, bn0 = bq0, bn1 = bq1, bn2 = bq2, bn3 = bq3;
      if (kt < 7) {
        const int k1 = (kt + 1) * 32;
        afn = load_a_frag_bf16(&hbuf[l15][0] + k1, hi);
        bn0 = load_b_frag(wrow + k1 + 0L * 256 * H, hi);
        bn1 = load_b_frag(wrow + k1 + 1L * 256 * H, hi);
        bn2 = load_b_frag(wrow + k1 + 2L * 256 * H, hi);
        bn3 = load_b_frag(wrow + k1 + 3L * 256 * H, hi);
      }
      acc[0] = __builtin_amdgcn_wmma_f32_16x16x32_bf16(
          false, af, false, bq0, (short)0, acc[0], false, false);
      acc[1] = __builtin_amdgcn_wmma_f32_16x16x32_bf16(
          false, af, false, bq1, (short)0, acc[1], false, false);
      acc[2] = __builtin_amdgcn_wmma_f32_16x16x32_bf16(
          false, af, false, bq2, (short)0, acc[2], false, false);
      acc[3] = __builtin_amdgcn_wmma_f32_16x16x32_bf16(
          false, af, false, bq3, (short)0, acc[3], false, false);
      af = afn; bq0 = bn0; bq1 = bn1; bq2 = bn2; bq3 = bn3;
    }

    // Drain this wave's async LDS writes, then make tile visible block-wide.
    asm volatile("s_wait_asynccnt 0x0" ::: "memory");
    __syncthreads();  // hbuf reads done; xgbuf(t) visible

#pragma unroll
    for (int r = 0; r < 8; ++r) {
      const int brow = hi * 8 + r;          // C/D row -> batch row in tile
      float iv = acc[0][r] + (float)xgbuf[brow][j];
      float fv = acc[1][r] + (float)xgbuf[brow][j + 256];
      float gv = acc[2][r] + (float)xgbuf[brow][j + 512];
      float ov = acc[3][r] + (float)xgbuf[brow][j + 768];
      float cn = fast_sigmoid(fv) * c[r] + fast_sigmoid(iv) * fast_tanh(gv);
      c[r] = cn;
      float hv = fast_sigmoid(ov) * fast_tanh(cn);
      bf16 hb = (bf16)hv;
      hbuf[brow][j] = hb;
      hout[((long)(b0 + brow) * S + t) * H + j] = hb;
    }
    __syncthreads();  // h(t) visible before next step's GEMM
  }
}

// ---------------------------------------------------------------------------
// Prep kernels
// ---------------------------------------------------------------------------
__global__ void cvt_f32_to_bf16(const float* __restrict__ in,
                                bf16* __restrict__ out, int n) {
  int i = blockIdx.x * blockDim.x + threadIdx.x;
  if (i < n) out[i] = (bf16)in[i];
}

__global__ void fuse_bias(const float* __restrict__ a,
                          const float* __restrict__ b,
                          float* __restrict__ out, int n) {
  int i = blockIdx.x * blockDim.x + threadIdx.x;
  if (i < n) out[i] = a[i] + b[i];
}

// ---------------------------------------------------------------------------
// Launch
// ---------------------------------------------------------------------------
extern "C" void kernel_launch(void* const* d_in, const int* in_sizes, int n_in,
                              void* d_out, int out_size, void* d_ws, size_t ws_size,
                              hipStream_t stream) {
  const float* x     = (const float*)d_in[0];
  const float* W_ih0 = (const float*)d_in[1];
  const float* W_hh0 = (const float*)d_in[2];
  const float* b_ih0 = (const float*)d_in[3];
  const float* b_hh0 = (const float*)d_in[4];
  const float* W_ih1 = (const float*)d_in[5];
  const float* W_hh1 = (const float*)d_in[6];
  const float* b_ih1 = (const float*)d_in[7];
  const float* b_hh1 = (const float*)d_in[8];
  const float* W_out = (const float*)d_in[9];
  const float* b_out = (const float*)d_in[10];
  float* out = (float*)d_out;

  // Workspace layout
  char* ws = (char*)d_ws;
  const size_t xg_bytes = (size_t)kM * kG * sizeof(bf16);   // 256 MB
  const size_t h_bytes  = (size_t)kM * kH * sizeof(bf16);   // 64 MB
  bf16* xgw = (bf16*)ws;
  bf16* hb  = (bf16*)(ws + xg_bytes);
  char* p   = ws + xg_bytes + h_bytes;
  bf16* wih0 = (bf16*)p; p += (size_t)kG * kV * sizeof(bf16);
  bf16* whh0 = (bf16*)p; p += (size_t)kG * kH * sizeof(bf16);
  bf16* wih1 = (bf16*)p; p += (size_t)kG * kH * sizeof(bf16);
  bf16* whh1 = (bf16*)p; p += (size_t)kG * kH * sizeof(bf16);
  bf16* wout = (bf16*)p; p += (size_t)kV * kH * sizeof(bf16);
  float* bias0 = (float*)p; p += (size_t)kG * sizeof(float);
  float* bias1 = (float*)p;

  // --- weight conversion + bias fusion -------------------------------------
  auto cvt = [&](const float* src, bf16* dst, int n) {
    cvt_f32_to_bf16<<<(n + 255) / 256, 256, 0, stream>>>(src, dst, n);
  };
  cvt(W_ih0, wih0, kG * kV);
  cvt(W_hh0, whh0, kG * kH);
  cvt(W_ih1, wih1, kG * kH);
  cvt(W_hh1, whh1, kG * kH);
  cvt(W_out, wout, kV * kH);
  fuse_bias<<<(kG + 255) / 256, 256, 0, stream>>>(b_ih0, b_hh0, bias0, kG);
  fuse_bias<<<(kG + 255) / 256, 256, 0, stream>>>(b_ih1, b_hh1, bias1, kG);

  const dim3 blk(256);
  const dim3 grid_g(kG / 128, (unsigned)(kM / 64));   // (8, 2048)
  const dim3 grid_o(kV / 128, (unsigned)(kM / 64));   // (1, 2048)

  // --- layer 0 -------------------------------------------------------------
  lstm_gemm_wmma<float, bf16><<<grid_g, blk, 0, stream>>>(
      x, wih0, bias0, xgw, kG, kV);
  lstm_scan_wmma<<<kB / 16, 512, 0, stream>>>(xgw, whh0, hb, kS);

  // --- layer 1 -------------------------------------------------------------
  lstm_gemm_wmma<bf16, bf16><<<grid_g, blk, 0, stream>>>(
      hb, wih1, bias1, xgw, kG, kH);
  lstm_scan_wmma<<<kB / 16, 512, 0, stream>>>(xgw, whh1, hb, kS);

  // --- output projection ---------------------------------------------------
  lstm_gemm_wmma<bf16, float><<<grid_o, blk, 0, stream>>>(
      hb, wout, b_out, out, kV, kH);
}
